// DistMultNN_62002147885503
// MI455X (gfx1250) — matrix operand order repaired and checked by reading, not compiled, not run on hardware
//
#include <hip/hip_runtime.h>
#include <stdint.h>

#define NDIM  256
#define BATCH 65536
#define TM    32          // samples per block

typedef __attribute__((ext_vector_type(16))) __bf16 v16bf;
typedef __attribute__((ext_vector_type(8)))  float  v8f;

union BfFrag { v16bf v; unsigned int u[8]; uint4 q[2]; };

// round-to-nearest-even f32 -> bf16, packed pair
__device__ __forceinline__ unsigned int pk_bf16(float lo, float hi) {
  unsigned int a = __float_as_uint(lo);
  unsigned int b = __float_as_uint(hi);
  a = (a + 0x7fffu + ((a >> 16) & 1u)) >> 16;
  b = (b + 0x7fffu + ((b >> 16) & 1u)) & 0xffff0000u;
  return b | a;
}

__device__ __forceinline__ unsigned short f2bf(float f) {
  unsigned int u = __float_as_uint(f);
  return (unsigned short)((u + 0x7fffu + ((u >> 16) & 1u)) >> 16);
}

// CDNA5 async memory->LDS copy (ASYNCcnt-tracked), GV addressing mode.
// ldsoff: wave-relative LDS byte offset (low 32 bits of generic pointer).
__device__ __forceinline__ void async_copy_b128(uint32_t ldsoff, const void* gaddr) {
  asm volatile("global_load_async_to_lds_b128 %0, %1, off"
               :: "v"(ldsoff), "v"((uint64_t)(uintptr_t)gaddr) : "memory");
}

__device__ __forceinline__ void wait_async0() {
#if __has_builtin(__builtin_amdgcn_s_wait_asynccnt)
  __builtin_amdgcn_s_wait_asynccnt(0);
#else
  asm volatile("s_wait_asynccnt 0" ::: "memory");
#endif
}

// A fragment (16x32 bf16) from an f32 row in LDS.
// 16-bit A layout: lanes 0-15 -> M=lane, K = k0+{0..7, 16..23};
//                  lanes16-31 -> M=lane-16, K = k0+{8..15, 24..31}.
__device__ __forceinline__ v16bf load_a_frag(const float* rowBase, int k0, int lane) {
  int koff = k0 + ((lane >> 4) << 3);
  float4 f0 = *(const float4*)(rowBase + koff);
  float4 f1 = *(const float4*)(rowBase + koff + 4);
  float4 f2 = *(const float4*)(rowBase + koff + 16);
  float4 f3 = *(const float4*)(rowBase + koff + 20);
  BfFrag A;
  A.u[0] = pk_bf16(f0.x, f0.y); A.u[1] = pk_bf16(f0.z, f0.w);
  A.u[2] = pk_bf16(f1.x, f1.y); A.u[3] = pk_bf16(f1.z, f1.w);
  A.u[4] = pk_bf16(f2.x, f2.y); A.u[5] = pk_bf16(f2.z, f2.w);
  A.u[6] = pk_bf16(f3.x, f3.y); A.u[7] = pk_bf16(f3.z, f3.w);
  return A.v;
}

// B fragment (32x16 bf16). B[k,n] = Bmat[n0+n][k] (Bmat row-major bf16:
// V rows for GEMM1 since B=V^T, W rows for GEMM2 by symmetry).
// Layout: lane holds column n=lane&15; lanes0-15 K=k0+{0..7,16..23}, lanes16-31 K=k0+{8..15,24..31}.
__device__ __forceinline__ v16bf load_b_frag(const unsigned short* Bmat, int n0, int k0, int lane) {
  int ncol = n0 + (lane & 15);
  int kb   = k0 + ((lane >> 4) << 3);
  const uint4* p = (const uint4*)(Bmat + ncol * NDIM + kb);
  BfFrag B;
  B.q[0] = p[0];
  B.q[1] = p[2];
  return B.v;
}

// ---------------------------------------------------------------------------
// prep: A = W - I; in-place Gauss-Jordan inverse (no pivoting; W-I well
// conditioned here); emit bf16 copies of V=inv(W-I) and W. Single block,
// CU-mode -> block-coherent through shared L0.
// ---------------------------------------------------------------------------
__global__ __launch_bounds__(1024) void prep_inverse_kernel(
    const float* __restrict__ W, float* __restrict__ A,
    unsigned short* __restrict__ Vbf, unsigned short* __restrict__ Wbf)
{
  const int tid = threadIdx.x;
  for (int i = tid; i < NDIM * NDIM; i += 1024) {
    int r = i >> 8, c = i & (NDIM - 1);
    float w = W[i];
    A[i] = w - (r == c ? 1.0f : 0.0f);
    Wbf[i] = f2bf(w);
  }
  __syncthreads();

  const int row = tid >> 2;            // 4 threads per row
  const int c0  = (tid & 3) << 6;      // 64 columns each
  for (int p = 0; p < NDIM; ++p) {
    float d   = 1.0f / A[p * NDIM + p];   // read before scaling
    float tip = A[row * NDIM + p];
    __syncthreads();
    if (row == p) {
      for (int j = c0; j < c0 + 64; ++j)
        A[p * NDIM + j] = (j == p) ? d : A[p * NDIM + j] * d;
    }
    __syncthreads();
    if (row != p) {
      for (int j = c0; j < c0 + 64; ++j) {
        float apj = A[p * NDIM + j];
        A[row * NDIM + j] = (j == p) ? (-tip * d) : (A[row * NDIM + j] - tip * apj);
      }
    }
    __syncthreads();
  }
  for (int i = tid; i < NDIM * NDIM; i += 1024)
    Vbf[i] = f2bf(A[i]);
}

// ---------------------------------------------------------------------------
// main fused kernel: 256 threads (8 wave32), 32 samples per block
// ---------------------------------------------------------------------------
__global__ __launch_bounds__(256) void hole_score_kernel(
    const float* __restrict__ Eent,
    const float* __restrict__ Erel,
    const float* __restrict__ bvec,
    const int*   __restrict__ samples,
    const unsigned short* __restrict__ Vbf,
    const unsigned short* __restrict__ Wbf,
    float* __restrict__ out)
{
  __shared__ float sE1[TM * NDIM];   // e1, later reused for h
  __shared__ float sE2[TM * NDIM];   // e2
  __shared__ float sY [TM * NDIM];   // y = r*cc - 0.5*b
  __shared__ float sB [NDIM];
  __shared__ float sQuad[TM];

  const int tid  = threadIdx.x;
  const int wave = tid >> 5;
  const int lane = tid & 31;
  const int s0   = blockIdx.x * TM;

  // ---- phase 1: async gather e1/e2 rows directly into LDS ----
  for (int it = tid; it < TM * (NDIM / 4); it += 256) {
    int s = it >> 6;
    int c = it & 63;
    int i1 = samples[(s0 + s) * 3 + 0];
    int i2 = samples[(s0 + s) * 3 + 2];
    async_copy_b128((uint32_t)(uintptr_t)&sE1[s * NDIM + 4 * c],
                    Eent + (size_t)i1 * NDIM + 4 * c);
    async_copy_b128((uint32_t)(uintptr_t)&sE2[s * NDIM + 4 * c],
                    Eent + (size_t)i2 * NDIM + 4 * c);
  }
  if (tid < NDIM / 4) *(float4*)&sB[4 * tid] = ((const float4*)bvec)[tid];
  if (tid < TM) sQuad[tid] = 0.0f;
  // warm L2 with the shared bf16 operand matrices while the gather flies
  __builtin_prefetch(Vbf + tid * 256, 0, 1);
  __builtin_prefetch(Wbf + tid * 256, 0, 1);
  wait_async0();
  __syncthreads();

  // ---- phase 2: circular correlation + y (VALU) ----
  for (int q = 0; q < TM / 8; ++q) {      // 4 samples per wave
    int s = wave * (TM / 8) + q;
    const float* e1 = &sE1[s * NDIM];
    const float* e2 = &sE2[s * NDIM];
    float acc[8];
#pragma unroll
    for (int t = 0; t < 8; ++t) acc[t] = 0.0f;
#pragma unroll 4
    for (int j = 0; j < NDIM; ++j) {
      float a = e1[j];                    // LDS broadcast
      int base = j + lane;
#pragma unroll
      for (int t = 0; t < 8; ++t)
        acc[t] += a * e2[(base + 32 * t) & (NDIM - 1)];
    }
    int ir = samples[(s0 + s) * 3 + 1];
    const float* rrow = Erel + (size_t)ir * NDIM;
#pragma unroll
    for (int t = 0; t < 8; ++t) {
      int k = lane + 32 * t;
      float x = rrow[k] * acc[t];
      sY[s * NDIM + k] = x - 0.5f * sB[k];
    }
  }
  __syncthreads();

  // wave -> output tile mapping: 2 m-tiles x 16 n-tiles, 4 n-tiles per wave
  const int m0    = (wave >> 2) << 4;     // 0 or 16
  const int nbase = (wave & 3) << 6;      // 0,64,128,192
  const int mrow  = m0 + (lane & 15);
  const int mb    = m0 + ((lane >> 4) << 3);

  // ---- phase 3: h = y @ V^T  (WMMA bf16, f32 accumulate) ----
  {
    v8f acc[4];
#pragma unroll
    for (int i = 0; i < 4; ++i)
#pragma unroll
      for (int v = 0; v < 8; ++v) acc[i][v] = 0.0f;

    for (int k0 = 0; k0 < NDIM; k0 += 32) {
      v16bf af = load_a_frag(&sY[mrow * NDIM], k0, lane);
#pragma unroll
      for (int i = 0; i < 4; ++i) {
        v16bf bfr = load_b_frag(Vbf, nbase + (i << 4), k0, lane);
        acc[i] = __builtin_amdgcn_wmma_f32_16x16x32_bf16(
            false, af, false, bfr, (short)0, acc[i], false, false);
      }
    }
    // store h (C/D layout: lanes0-15 M=v, lanes16-31 M=v+8; N=lane&15)
#pragma unroll
    for (int i = 0; i < 4; ++i) {
      int n = nbase + (i << 4) + (lane & 15);
#pragma unroll
      for (int v = 0; v < 8; ++v)
        sE1[(mb + v) * NDIM + n] = acc[i][v];
    }
  }
  __syncthreads();

  // ---- phase 4: quad[m] = h W h via G = h @ W (W symmetric), fused reduce ----
  {
    v8f acc[4];
#pragma unroll
    for (int i = 0; i < 4; ++i)
#pragma unroll
      for (int v = 0; v < 8; ++v) acc[i][v] = 0.0f;

    for (int k0 = 0; k0 < NDIM; k0 += 32) {
      v16bf af = load_a_frag(&sE1[mrow * NDIM], k0, lane);
#pragma unroll
      for (int i = 0; i < 4; ++i) {
        v16bf bfr = load_b_frag(Wbf, nbase + (i << 4), k0, lane);
        acc[i] = __builtin_amdgcn_wmma_f32_16x16x32_bf16(
            false, af, false, bfr, (short)0, acc[i], false, false);
      }
    }
#pragma unroll
    for (int i = 0; i < 4; ++i) {
      int n = nbase + (i << 4) + (lane & 15);
#pragma unroll
      for (int v = 0; v < 8; ++v) {
        float val = acc[i][v] * sE1[(mb + v) * NDIM + n];  // G[m,n]*h[m,n]
#pragma unroll
        for (int off = 1; off < 16; off <<= 1)             // reduce over n half
          val += __shfl_xor(val, off, 32);
        if ((lane & 15) == 0) atomicAdd(&sQuad[mb + v], val);
      }
    }
  }
  __syncthreads();

  // ---- phase 5: out = quad + h.b - ||h - x||^2  (x = y + 0.5 b) ----
  {
    int s  = tid >> 3;                    // 8 threads per sample
    int kq = tid & 7;
    const float* hrow = &sE1[s * NDIM];
    const float* yrow = &sY[s * NDIM];
    float part = 0.0f;
#pragma unroll 4
    for (int k = kq * 32; k < kq * 32 + 32; ++k) {
      float h  = hrow[k];
      float x  = yrow[k] + 0.5f * sB[k];
      float dd = h - x;
      part += h * sB[k] - dd * dd;
    }
#pragma unroll
    for (int off = 1; off < 8; off <<= 1)
      part += __shfl_xor(part, off, 32);
    if (kq == 0) out[s0 + s] = part + sQuad[s];
  }
}

extern "C" void kernel_launch(void* const* d_in, const int* in_sizes, int n_in,
                              void* d_out, int out_size, void* d_ws, size_t ws_size,
                              hipStream_t stream) {
  const float* Eent    = (const float*)d_in[0];
  const float* Erel    = (const float*)d_in[1];
  const float* W       = (const float*)d_in[2];
  const float* bvec    = (const float*)d_in[3];
  const int*   samples = (const int*)d_in[4];
  float* out = (float*)d_out;

  // workspace layout: [0,256K) f32 scratch A; [256K,384K) bf16 V; [384K,512K) bf16 W
  float* A = (float*)d_ws;
  unsigned short* Vbf = (unsigned short*)((char*)d_ws + 256 * 1024);
  unsigned short* Wbf = (unsigned short*)((char*)d_ws + 384 * 1024);

  hipLaunchKernelGGL(prep_inverse_kernel, dim3(1), dim3(1024), 0, stream,
                     W, A, Vbf, Wbf);
  hipLaunchKernelGGL(hole_score_kernel, dim3(BATCH / TM), dim3(256), 0, stream,
                     Eent, Erel, bvec, samples, Vbf, Wbf, out);
}
